// GCN_40424232190064
// MI455X (gfx1250) — matrix operand order repaired and checked by reading, not compiled, not run on hardware
//
#include <hip/hip_runtime.h>

// ---------- types / helpers ----------
typedef __attribute__((ext_vector_type(16))) __bf16 v16bf;
typedef __attribute__((ext_vector_type(8)))  float  v8f;

__device__ __forceinline__ __bf16 f2b(float f) {
  union { float f; unsigned u; } x; x.f = f;
  unsigned r = x.u + 0x7FFFu + ((x.u >> 16) & 1u);   // round-to-nearest-even
  unsigned short hs = (unsigned short)(r >> 16);
  union { unsigned short s; __bf16 b; } y; y.s = hs; return y.b;
}
__device__ __forceinline__ float b2f(__bf16 b) {
  union { __bf16 b; unsigned short s; } y; y.b = b;
  union { unsigned u; float f; } x; x.u = ((unsigned)y.s) << 16;
  return x.f;
}

// ---------- pack W[K,NOUT] (fp32) into bf16 hi/lo B-fragments ----------
// Fragment layout (ISA 7.12.2, 16-bit B 32x16): element j of lane L for slab kt,
// tile nt holds W[kt*32 + (j<8?0:16) + (L>>4)*8 + (j&7)][nt*16 + (L&15)], zero-padded OOB.
// Storage: ((kt*NT + nt)*32 + lane)*16 + j   -> wave loads one v16bf per fragment.
template<int K, int NT, int NOUT>
__global__ void pack_b_kernel(const float* __restrict__ W,
                              __bf16* __restrict__ Wh, __bf16* __restrict__ Wl) {
  const int KT = (K + 31) / 32;
  long t = (long)blockIdx.x * blockDim.x + threadIdx.x;
  long total = (long)KT * NT * 32 * 16;
  if (t >= total) return;
  int j    = (int)(t & 15);
  int lane = (int)((t >> 4) & 31);
  long rest = t >> 9;
  int nt = (int)(rest % NT);
  int kt = (int)(rest / NT);
  int k = kt * 32 + ((j < 8) ? 0 : 16) + (lane >> 4) * 8 + (j & 7);
  int n = nt * 16 + (lane & 15);
  float v = (k < K && n < NOUT) ? W[k * NOUT + n] : 0.0f;
  __bf16 h = f2b(v);
  Wh[t] = h;
  Wl[t] = f2b(v - b2f(h));
}

// ---------- split-bf16 WMMA GEMM:  C[M,NOUT] = A[M,K] @ B[K,NOUT] ----------
// One wave per 16-row tile; acc += Ah*Bh + Ah*Bl + Al*Bh (fp32 accumulate).
// Main loop: guard-free float4 A loads + packed v16bf B loads; K%32 tail once.
template<int K, int NT, int NOUT>
__global__ __launch_bounds__(256) void gemm_bf16x3(const float* __restrict__ A,
                                                   const __bf16* __restrict__ Bh,
                                                   const __bf16* __restrict__ Bl,
                                                   float* __restrict__ C, int M) {
  constexpr int KFULL = K / 32;
  constexpr bool TAIL = (K % 32) != 0;
  int wave = blockIdx.x * 8 + (threadIdx.x >> 5);
  int lane = threadIdx.x & 31;
  int m0 = wave * 16;
  if (m0 >= M) return;                 // wave-uniform: EXEC all-ones around WMMA
  int half = lane >> 4;
  int l15  = lane & 15;
  const float* arow = A + (long)(m0 + l15) * K;
  const v16bf* BH = (const v16bf*)Bh;
  const v16bf* BL = (const v16bf*)Bl;

  v8f acc[NT];
  v8f vzero = {};
#pragma unroll
  for (int t = 0; t < NT; ++t) acc[t] = vzero;

  for (int kt = 0; kt < KFULL; ++kt) {
    int kb = kt * 32 + half * 8;
    __builtin_prefetch(arow + kt * 32 + 64, 0, 1);
    // A fragment: j=0..7 -> k=kb+j ; j=8..15 -> k=kb+16+(j-8); all 16B aligned
    float4 a0 = *(const float4*)(arow + kb);
    float4 a1 = *(const float4*)(arow + kb + 4);
    float4 a2 = *(const float4*)(arow + kb + 16);
    float4 a3 = *(const float4*)(arow + kb + 20);
    float av[16] = {a0.x, a0.y, a0.z, a0.w, a1.x, a1.y, a1.z, a1.w,
                    a2.x, a2.y, a2.z, a2.w, a3.x, a3.y, a3.z, a3.w};
    v16bf ah, al;
#pragma unroll
    for (int j = 0; j < 16; ++j) {
      __bf16 h = f2b(av[j]);
      ah[j] = h;
      al[j] = f2b(av[j] - b2f(h));
    }
#pragma unroll
    for (int t = 0; t < NT; ++t) {
      long fi = (long)(kt * NT + t) * 32 + lane;
      v16bf bh = BH[fi];
      v16bf bl = BL[fi];
      acc[t] = __builtin_amdgcn_wmma_f32_16x16x32_bf16(false, ah, false, bh, (short)0, acc[t], false, false);
      acc[t] = __builtin_amdgcn_wmma_f32_16x16x32_bf16(false, ah, false, bl, (short)0, acc[t], false, false);
      acc[t] = __builtin_amdgcn_wmma_f32_16x16x32_bf16(false, al, false, bh, (short)0, acc[t], false, false);
    }
  }

  if constexpr (TAIL) {                // single remainder slab (e.g. k=480..499)
    int kbase = KFULL * 32;
    v16bf ah, al;
#pragma unroll
    for (int j = 0; j < 16; ++j) {
      int k = kbase + ((j < 8) ? 0 : 16) + half * 8 + (j & 7);
      float v = (k < K) ? arow[k] : 0.0f;
      __bf16 h = f2b(v);
      ah[j] = h;
      al[j] = f2b(v - b2f(h));
    }
#pragma unroll
    for (int t = 0; t < NT; ++t) {
      long fi = (long)(KFULL * NT + t) * 32 + lane;
      v16bf bh = BH[fi];
      v16bf bl = BL[fi];
      acc[t] = __builtin_amdgcn_wmma_f32_16x16x32_bf16(false, ah, false, bh, (short)0, acc[t], false, false);
      acc[t] = __builtin_amdgcn_wmma_f32_16x16x32_bf16(false, ah, false, bl, (short)0, acc[t], false, false);
      acc[t] = __builtin_amdgcn_wmma_f32_16x16x32_bf16(false, al, false, bh, (short)0, acc[t], false, false);
    }
  }

  // C/D layout: VGPR v -> row m0 + 8*half + v, col = nt*16 + (lane&15)
#pragma unroll
  for (int t = 0; t < NT; ++t) {
    int n = t * 16 + l15;
    if (n < NOUT) {
#pragma unroll
      for (int v = 0; v < 8; ++v) {
        int row = m0 + half * 8 + v;
        C[(long)row * NOUT + n] = acc[t][v];
      }
    }
  }
}

// ---------- graph kernels ----------
__global__ void fill_kernel(float* __restrict__ p, float val, long n) {
  long i = (long)blockIdx.x * blockDim.x + threadIdx.x;
  if (i < n) p[i] = val;
}

__global__ void deg_kernel(const int* __restrict__ dst, float* __restrict__ deg, long E) {
  long i = (long)blockIdx.x * blockDim.x + threadIdx.x;
  if (i < E) unsafeAtomicAdd(&deg[dst[i]], 1.0f);
}

__global__ void rsqrt_kernel(float* __restrict__ d, int N) {
  int i = blockIdx.x * blockDim.x + threadIdx.x;
  if (i < N) d[i] = rsqrtf(d[i]);      // deg >= 1 (self loop), no zero guard needed
}

// per-edge scatter-add: out[dst] += feat[src] * dinv[src]*dinv[dst]
template<int F>
__global__ void scatter_kernel(const float* __restrict__ feat, const int* __restrict__ src,
                               const int* __restrict__ dst, const float* __restrict__ dinv,
                               float* __restrict__ out, long E) {
  const int C4 = F / 4;
  long t = (long)blockIdx.x * blockDim.x + threadIdx.x;
  if (t >= E * C4) return;
  long e = t / C4;
  int  c = (int)(t - e * C4);
  int s = src[e], d = dst[e];
  float nrm = dinv[s] * dinv[d];
  const float4 v = *(const float4*)(feat + (long)s * F + c * 4);
  float* o = out + (long)d * F + c * 4;
  unsafeAtomicAdd(o + 0, v.x * nrm);
  unsafeAtomicAdd(o + 1, v.y * nrm);
  unsafeAtomicAdd(o + 2, v.z * nrm);
  unsafeAtomicAdd(o + 3, v.w * nrm);
}

// h = relu(h + xw * dinv^2 + b)   (self-loop term + bias, F = 64)
__global__ void finalize_relu_kernel(float* __restrict__ h, const float* __restrict__ xw,
                                     const float* __restrict__ dinv, const float* __restrict__ b,
                                     long n) {
  long idx = (long)blockIdx.x * blockDim.x + threadIdx.x;
  if (idx >= n) return;
  long i = idx >> 6; int f = (int)(idx & 63);
  float di = dinv[i];
  float v = h[idx] + xw[idx] * di * di + b[f];
  h[idx] = v > 0.0f ? v : 0.0f;
}

// logits += hw * dinv^2 + b   (F = 40)
__global__ void finalize2_kernel(float* __restrict__ logits, const float* __restrict__ hw,
                                 const float* __restrict__ dinv, const float* __restrict__ b,
                                 long n) {
  long idx = (long)blockIdx.x * blockDim.x + threadIdx.x;
  if (idx >= n) return;
  long i = idx / 40; int f = (int)(idx - i * 40);
  float di = dinv[i];
  logits[idx] += hw[idx] * di * di + b[f];
}

// one wave32 per row of 40 classes; lane covers c and c+32
__global__ void logsoftmax_kernel(const float* __restrict__ logits, float* __restrict__ out, int N) {
  int warp = blockIdx.x * 8 + (threadIdx.x >> 5);
  int lane = threadIdx.x & 31;
  if (warp >= N) return;
  const float* row = logits + (long)warp * 40;
  float v0 = row[lane];
  float v1 = (lane < 8) ? row[32 + lane] : -3.0e38f;
  float m = fmaxf(v0, v1);
#pragma unroll
  for (int off = 16; off > 0; off >>= 1) m = fmaxf(m, __shfl_xor(m, off, 32));
  float s = __expf(v0 - m) + ((lane < 8) ? __expf(v1 - m) : 0.0f);
#pragma unroll
  for (int off = 16; off > 0; off >>= 1) s += __shfl_xor(s, off, 32);
  float lse = m + logf(s);
  float* orow = out + (long)warp * 40;
  orow[lane] = v0 - lse;
  if (lane < 8) orow[32 + lane] = v1 - lse;
}

// ---------- launcher ----------
extern "C" void kernel_launch(void* const* d_in, const int* in_sizes, int n_in,
                              void* d_out, int out_size, void* d_ws, size_t ws_size,
                              hipStream_t stream) {
  const float* X  = (const float*)d_in[0];
  const int*   EI = (const int*)d_in[1];
  const float* W1 = (const float*)d_in[2];
  const float* B1 = (const float*)d_in[3];
  const float* W2 = (const float*)d_in[4];
  const float* B2 = (const float*)d_in[5];

  const int  NF = 500, NH = 64, NC = 40;
  const int  N  = in_sizes[0] / NF;          // 100000
  const long E  = (long)in_sizes[1] / 2;     // 1600000
  const int* SRC = EI;
  const int* DST = EI + E;

  // workspace layout (floats): dinv | xw[N*64] | h[N*64] | hw2[N*40] | packed W
  float* ws   = (float*)d_ws;
  long   o0   = ((long)N + 127) & ~127L;
  float* dinv = ws;
  float* xw   = ws + o0;
  float* h    = xw + (long)N * NH;
  float* hw2  = h  + (long)N * NH;

  const long W1FRAG = 16L * 4 * 32 * 16;     // KT=16, NT=4  -> 32768 bf16
  const long W2FRAG = 2L * 3 * 32 * 16;      // KT=2,  NT=3  -> 3072 bf16
  __bf16* w1h = (__bf16*)(hw2 + (long)N * NC);
  __bf16* w1l = w1h + W1FRAG;
  __bf16* w2h = w1l + W1FRAG;
  __bf16* w2l = w2h + W2FRAG;

  float* lsm    = (float*)d_out;                    // log_softmax output
  float* logits = (float*)d_out + (long)N * NC;     // logits output (built in place)

  const int T = 256;

  // degrees (self-loop folded in as init 1.0), then dinv = rsqrt(deg)
  fill_kernel<<<(N + T - 1) / T, T, 0, stream>>>(dinv, 1.0f, (long)N);
  deg_kernel<<<(int)((E + T - 1) / T), T, 0, stream>>>(DST, dinv, E);
  rsqrt_kernel<<<(N + T - 1) / T, T, 0, stream>>>(dinv, N);

  // pack weights into bf16 hi/lo fragments (tiny, once per call)
  pack_b_kernel<500, 4, 64><<<(int)((W1FRAG + T - 1) / T), T, 0, stream>>>(W1, w1h, w1l);
  pack_b_kernel<64, 3, 40><<<(int)((W2FRAG + T - 1) / T), T, 0, stream>>>(W2, w2h, w2l);

  // layer 1: xw = X @ W1   (split-bf16 WMMA, K=500, N=64)
  int tiles = (N + 15) / 16;
  int gblk  = (tiles + 7) / 8;
  gemm_bf16x3<500, 4, 64><<<gblk, 256, 0, stream>>>(X, w1h, w1l, xw, N);

  long hn = (long)N * NH;
  fill_kernel<<<(int)((hn + T - 1) / T), T, 0, stream>>>(h, 0.0f, hn);
  scatter_kernel<64><<<(int)((E * 16 + T - 1) / T), T, 0, stream>>>(xw, SRC, DST, dinv, h, E);
  finalize_relu_kernel<<<(int)((hn + T - 1) / T), T, 0, stream>>>(h, xw, dinv, B1, hn);

  // layer 2: hw2 = h @ W2   (K=64, N=40 -> 3 padded n-tiles)
  gemm_bf16x3<64, 3, 40><<<gblk, 256, 0, stream>>>(h, w2h, w2l, hw2, N);

  long ln = (long)N * NC;
  fill_kernel<<<(int)((ln + T - 1) / T), T, 0, stream>>>(logits, 0.0f, ln);
  scatter_kernel<40><<<(int)((E * 10 + T - 1) / T), T, 0, stream>>>(hw2, SRC, DST, dinv, logits, E);
  finalize2_kernel<<<(int)((ln + T - 1) / T), T, 0, stream>>>(logits, hw2, dinv, B2, ln);

  // log_softmax into first half of d_out
  logsoftmax_kernel<<<(N + 7) / 8, 256, 0, stream>>>(logits, lsm, N);
}
